// SelfAttention_9388798509737
// MI455X (gfx1250) — matrix operand order repaired and checked by reading, not compiled, not run on hardware
//
#include <hip/hip_runtime.h>
#include <math.h>

// ---------------- problem constants (fixed by setup_inputs) ----------------
#define N_PTS 256
#define C_DIM 128
#define KNN   10
#define P_ROWS (N_PTS*KNN)          // 2560 graph-feature rows
#define NEG_SLOPE 0.2f
#define EPS_F 1e-5f
#define H_HEADS 4
#define CH 32                        // C / H
#define INV_SQRT_CH 0.17677669529663687f
#define NEG_LOG_1E4 -9.210340371976184f

typedef __attribute__((ext_vector_type(16))) _Float16 v16h;
typedef __attribute__((ext_vector_type(8)))  _Float16 v8h;
typedef __attribute__((ext_vector_type(8)))  float    v8f;

union ABfrag { v16h v; v8h h[2]; };

// ---------------------------------------------------------------------------
// Simple f16->f32 WMMA GEMM:  C[M,N] = A[M,K] * Bt[N,K]^T (+ bias[N])
// One 16x16 tile per wave; used for odd shapes (N=32 att@V, K=512 conv3).
// Fragment k order: contiguous, k = 16*(lane/16) + s, same in A and B (16-bit A 16x32; B K-split by lane half).
// ---------------------------------------------------------------------------
__global__ void __launch_bounds__(256) gemm_f16_wmma(
    const _Float16* __restrict__ A, const _Float16* __restrict__ Bt,
    const float* __restrict__ bias, float* __restrict__ C,
    int M, int N, int K, int ldc,
    long sA, long sB, long sC)
{
  int b = blockIdx.y;
  A  += (long)b * sA;  Bt += (long)b * sB;  C += (long)b * sC;

  int wave = blockIdx.x * (blockDim.x >> 5) + (threadIdx.x >> 5);
  int lane = threadIdx.x & 31;
  int ntiles = N >> 4;
  int tiles  = (M >> 4) * ntiles;
  if (wave >= tiles) return;                 // wave-uniform: EXEC stays all-1s for WMMA

  int mt = wave / ntiles, nt = wave % ntiles;
  int l  = lane & 15, hf = lane >> 4;
  int row = mt * 16 + l;
  int col = nt * 16 + l;

  const _Float16* arow = A  + (long)row * K;
  const _Float16* brow = Bt + (long)col * K;

  v8f acc = {};
  for (int k0 = 0; k0 < K; k0 += 32) {
    ABfrag a, bf;
    a.h[0]  = *(const v8h*)(arow + k0 + hf * 16);
    a.h[1]  = *(const v8h*)(arow + k0 + hf * 16 + 8);
    bf.h[0] = *(const v8h*)(brow + k0 + hf * 16);
    bf.h[1] = *(const v8h*)(brow + k0 + hf * 16 + 8);
    acc = __builtin_amdgcn_wmma_f32_16x16x32_f16(
        false, a.v, false, bf.v, (short)0, acc, false, false);
  }

  float bv = bias ? bias[col] : 0.f;
  #pragma unroll
  for (int r = 0; r < 8; ++r) {
    int orow = mt * 16 + r + hf * 8;
    C[(long)orow * ldc + col] = acc[r] + bv;
  }
}

// ---------------------------------------------------------------------------
// Blocked WMMA GEMM with async-to-LDS B-panel staging (CDNA5 ASYNCcnt path).
// Block = 8 waves computes a 128x64 macro-tile. The 64xK B panel (contiguous,
// since Bt is stored [col][K]) is staged into LDS with
// GLOBAL_LOAD_ASYNC_TO_LDS_B128, waited with s_wait_asynccnt, then each wave
// reuses its A fragment across 4 WMMAs reading B fragments via ds_load_b128.
// Requires: M%128==0, N%64==0, K%32==0, K<=256 (panel <= 32KB LDS).
// ---------------------------------------------------------------------------
__global__ void __launch_bounds__(256) gemm_f16_wmma_blk(
    const _Float16* __restrict__ A, const _Float16* __restrict__ Bt,
    const float* __restrict__ bias, float* __restrict__ C,
    int M, int N, int K, int ldc,
    long sA, long sB, long sC)
{
  extern __shared__ _Float16 Bs[];       // only LDS object -> starts at offset 0
  int b = blockIdx.y;
  A  += (long)b * sA;  Bt += (long)b * sB;  C += (long)b * sC;

  int nblks = N >> 6;
  int mb = blockIdx.x / nblks;
  int nb = blockIdx.x % nblks;

  // ---- cooperative async stage of the contiguous 64*K*2-byte B panel ----
  {
    const char* gsrc = (const char*)(Bt + (size_t)nb * 64 * K);
    int chunks = (K * 128) >> 4;                 // (64*K*2)/16 bytes -> b128 chunks
    for (int ch = threadIdx.x; ch < chunks; ch += 256) {   // uniform trip count
      unsigned           lo = (unsigned)(ch << 4);         // LDS byte offset
      unsigned long long ga = (unsigned long long)gsrc + ((unsigned long long)ch << 4);
      asm volatile("global_load_async_to_lds_b128 %0, %1, off"
                   :: "v"(lo), "v"(ga) : "memory");
    }
    asm volatile("s_wait_asynccnt 0x0" ::: "memory");
    __syncthreads();
  }

  int wave = threadIdx.x >> 5;
  int lane = threadIdx.x & 31;
  int l = lane & 15, hf = lane >> 4;
  int row = mb * 128 + wave * 16 + l;
  const _Float16* arow = A + (long)row * K;

  v8f acc[4] = {};
  for (int k0 = 0; k0 < K; k0 += 32) {
    ABfrag a;
    a.h[0] = *(const v8h*)(arow + k0 + hf * 16);
    a.h[1] = *(const v8h*)(arow + k0 + hf * 16 + 8);
    #pragma unroll
    for (int j = 0; j < 4; ++j) {
      ABfrag bf;
      const _Float16* bl = Bs + (size_t)(j * 16 + l) * K + k0 + hf * 16;
      bf.h[0] = *(const v8h*)bl;
      bf.h[1] = *(const v8h*)(bl + 8);
      acc[j] = __builtin_amdgcn_wmma_f32_16x16x32_f16(
          false, a.v, false, bf.v, (short)0, acc[j], false, false);
    }
  }

  #pragma unroll
  for (int j = 0; j < 4; ++j) {
    int col = nb * 64 + j * 16 + l;
    float bv = bias ? bias[col] : 0.f;
    #pragma unroll
    for (int r = 0; r < 8; ++r) {
      int orow = mb * 128 + wave * 16 + r + hf * 8;
      C[(long)orow * ldc + col] = acc[j][r] + bv;
    }
  }
}

// ---------------------------------------------------------------------------
// kNN: one thread per point, coords staged in LDS, insertion-sorted top-(k+1),
// drop self (nearest). Strict '<' keeps lower index first on ties (top_k order).
// ---------------------------------------------------------------------------
__global__ void __launch_bounds__(256) knn_kernel(const float* __restrict__ coords,
                                                  int* __restrict__ idx)
{
  __shared__ float px[N_PTS], py[N_PTS], pz[N_PTS];
  int t = threadIdx.x;
  px[t] = coords[t];
  py[t] = coords[N_PTS + t];
  pz[t] = coords[2 * N_PTS + t];
  __syncthreads();

  float bd[KNN + 1];
  int   bi[KNN + 1];
  #pragma unroll
  for (int j = 0; j <= KNN; ++j) { bd[j] = 3.4e38f; bi[j] = -1; }

  float x = px[t], y = py[t], z = pz[t];
  for (int m = 0; m < N_PTS; ++m) {
    float dx = px[m] - x, dy = py[m] - y, dz = pz[m] - z;
    float d = fmaxf(dx * dx + dy * dy + dz * dz, 1e-12f);
    if (d < bd[KNN]) {
      int pos = KNN;
      while (pos > 0 && d < bd[pos - 1]) {
        bd[pos] = bd[pos - 1]; bi[pos] = bi[pos - 1]; --pos;
      }
      bd[pos] = d; bi[pos] = m;
    }
  }
  for (int j = 0; j < KNN; ++j) idx[t * KNN + j] = bi[j + 1];
}

// ------------------------- small conversion helpers -------------------------
__global__ void f2h_kernel(const float* __restrict__ s, _Float16* __restrict__ d, int n)
{
  int g = blockIdx.x * blockDim.x + threadIdx.x;
  if (g < n) d[g] = (_Float16)s[g];
}

// dst[j*R + i] = src[i*Cc + j]  (store W^T so GEMM B loads are contiguous in K)
__global__ void f2hT_kernel(const float* __restrict__ s, _Float16* __restrict__ d, int R, int Cc)
{
  int g = blockIdx.x * blockDim.x + threadIdx.x;
  if (g >= R * Cc) return;
  int j = g / R, i = g % R;
  d[g] = (_Float16)s[i * Cc + j];
}

// graph feature rows (shared by conv1 and conv2 stages; feats is [Cin][N]):
// A[p][c] = c<Cin ? f[c][n] : f[c-Cin][idx[p]] - f[c-Cin][n]
__global__ void prep_gf_kernel(const float* __restrict__ feats, const int* __restrict__ idx,
                               _Float16* __restrict__ A, int Cin)
{
  int g = blockIdx.x * blockDim.x + threadIdx.x;
  int tot = P_ROWS * 2 * Cin;
  if (g >= tot) return;
  int c = g % (2 * Cin);
  int p = g / (2 * Cin);
  int n = p / KNN;
  float v;
  if (c < Cin) v = feats[c * N_PTS + n];
  else { int cc = c - Cin; int j = idx[p]; v = feats[cc * N_PTS + j] - feats[cc * N_PTS + n]; }
  A[g] = (_Float16)v;
}

// x3 input: concat [features(C), x1(C), x2(2C)] -> [n][512] f16
__global__ void prep_cat_kernel(const float* __restrict__ f, const float* __restrict__ x1,
                                const float* __restrict__ x2, _Float16* __restrict__ A)
{
  int g = blockIdx.x * blockDim.x + threadIdx.x;
  if (g >= N_PTS * 512) return;
  int c = g & 511, n = g >> 9;
  float v;
  if (c < 128)       v = f[c * N_PTS + n];
  else if (c < 256)  v = x1[(c - 128) * N_PTS + n];
  else               v = x2[(c - 256) * N_PTS + n];
  A[g] = (_Float16)v;
}

// InstanceNorm over (N,k) per channel + LeakyReLU + max over k -> xo[c][n]
__global__ void __launch_bounds__(256) inorm_max_kernel(const float* __restrict__ y,
                                                        float* __restrict__ xo, int O)
{
  __shared__ float s1[256], s2[256];
  int c = blockIdx.x, t = threadIdx.x;
  float a = 0.f, b = 0.f;
  for (int p = t; p < P_ROWS; p += 256) {
    float v = y[(long)p * O + c]; a += v; b += v * v;
  }
  s1[t] = a; s2[t] = b; __syncthreads();
  for (int o = 128; o > 0; o >>= 1) {
    if (t < o) { s1[t] += s1[t + o]; s2[t] += s2[t + o]; }
    __syncthreads();
  }
  float mean = s1[0] / (float)P_ROWS;
  float var  = s2[0] / (float)P_ROWS - mean * mean;
  float rstd = rsqrtf(var + EPS_F);
  float mx = -3.4e38f;
  for (int kk = 0; kk < KNN; ++kk) {
    float v = (y[(long)(t * KNN + kk) * O + c] - mean) * rstd;
    v = v >= 0.f ? v : NEG_SLOPE * v;
    mx = fmaxf(mx, v);
  }
  xo[c * N_PTS + t] = mx;
}

// InstanceNorm over N per channel + LeakyReLU -> x3[n][c] (f32 + f16 copy)
__global__ void __launch_bounds__(256) inorm_nl_kernel(const float* __restrict__ y,
                                                       float* __restrict__ x3,
                                                       _Float16* __restrict__ x3h)
{
  __shared__ float s1[256], s2[256];
  int c = blockIdx.x, t = threadIdx.x;   // t = n
  float v = y[t * C_DIM + c];
  s1[t] = v; s2[t] = v * v; __syncthreads();
  for (int o = 128; o > 0; o >>= 1) {
    if (t < o) { s1[t] += s1[t + o]; s2[t] += s2[t + o]; }
    __syncthreads();
  }
  float mean = s1[0] / (float)N_PTS;
  float var  = s2[0] / (float)N_PTS - mean * mean;
  float w = (v - mean) * rsqrtf(var + EPS_F);
  w = w >= 0.f ? w : NEG_SLOPE * w;
  x3[t * C_DIM + c]  = w;
  x3h[t * C_DIM + c] = (_Float16)w;
}

// gbp[j] = bp[j] + sum_o proj1_b[o] * wp[o][j]   (bias folding for sp)
__global__ void gbp_kernel(const float* __restrict__ p1b, const float* __restrict__ wp,
                           const float* __restrict__ bp, float* __restrict__ gbp)
{
  int j = threadIdx.x;
  float s = bp[j];
  for (int o = 0; o < C_DIM; ++o) s += p1b[o] * wp[o * C_DIM + j];
  gbp[j] = s;
}

// rr[n][h][e] = sum_{jj<32} G[e][h*32+jj] * q[n][h*32+jj]
__global__ void __launch_bounds__(128) rr_kernel(const float* __restrict__ G,
                                                 const float* __restrict__ q,
                                                 float* __restrict__ rr)
{
  __shared__ float qs[C_DIM];
  int n = blockIdx.x, e = threadIdx.x;
  qs[e] = q[n * C_DIM + e];
  __syncthreads();
  float acc[H_HEADS] = {0.f, 0.f, 0.f, 0.f};
  for (int jj = 0; jj < CH; ++jj) {
    #pragma unroll
    for (int h = 0; h < H_HEADS; ++h)
      acc[h] += G[e * C_DIM + h * CH + jj] * qs[h * CH + jj];
  }
  #pragma unroll
  for (int h = 0; h < H_HEADS; ++h)
    rr[((n * H_HEADS) + h) * C_DIM + e] = acc[h];
}

// Fused score kernel: one wave per (n,m).
// scores[h][n][m] = ( q.k  +  q.(gb+bp)  +  mean_k( sinemb(ang) . rr ) ) / sqrt(ch)
__global__ void __launch_bounds__(256) score_kernel(
    const float* __restrict__ coords, const int* __restrict__ idx,
    const float* __restrict__ q, const float* __restrict__ kmat,
    const float* __restrict__ rr, const float* __restrict__ gbp,
    float* __restrict__ scores)
{
  int wave = blockIdx.x * 8 + (threadIdx.x >> 5);
  int lane = threadIdx.x & 31;
  int n = wave >> 8, m = wave & 255;

  float xn = coords[n], yn = coords[N_PTS + n], zn = coords[2 * N_PTS + n];
  float ax = coords[m] - xn, ay = coords[N_PTS + m] - yn, az = coords[2 * N_PTS + m] - zn;
  float ma = sqrtf(ax * ax + ay * ay + az * az);

  float ang = 0.f;
  if (lane < KNN) {
    int j = idx[n * KNN + lane];
    float rx = coords[j] - xn, ry = coords[N_PTS + j] - yn, rz = coords[2 * N_PTS + j] - zn;
    float dot = rx * ax + ry * ay + rz * az;
    float mr  = sqrtf(rx * rx + ry * ry + rz * rz);
    ang = atan2f(dot, mr * ma) * 0.5f;
  }

  // lane owns freq pairs t = lane and t = lane+32  (div = exp(2t * -ln(1e4)/128))
  float d0 = __expf((float)(2 * lane)      * (NEG_LOG_1E4 / 128.f));
  float d1 = __expf((float)(2 * lane + 64) * (NEG_LOG_1E4 / 128.f));
  float ss0 = 0.f, cc0 = 0.f, ss1 = 0.f, cc1 = 0.f;
  #pragma unroll
  for (int kk = 0; kk < KNN; ++kk) {
    float a = __shfl(ang, kk, 32);
    float s, c;
    __sincosf(a * d0, &s, &c); ss0 += s; cc0 += c;
    __sincosf(a * d1, &s, &c); ss1 += s; cc1 += c;
  }

  float sp[H_HEADS];
  #pragma unroll
  for (int h = 0; h < H_HEADS; ++h) {
    const float* r = rr + ((n * H_HEADS) + h) * C_DIM;
    sp[h] = ss0 * r[2 * lane]      + cc0 * r[2 * lane + 1]
          + ss1 * r[2 * lane + 64] + cc1 * r[2 * lane + 65];
  }

  int c0 = lane * 4;
  float se = 0.f, cq = 0.f;
  #pragma unroll
  for (int u = 0; u < 4; ++u) {
    float qv = q[n * C_DIM + c0 + u];
    se += qv * kmat[m * C_DIM + c0 + u];
    cq += qv * gbp[c0 + u];
  }
  int myh = lane >> 3;
  float uu[H_HEADS];
  #pragma unroll
  for (int h = 0; h < H_HEADS; ++h) uu[h] = (h == myh) ? (se + cq) : 0.f;

  #pragma unroll
  for (int off = 16; off > 0; off >>= 1) {
    #pragma unroll
    for (int h = 0; h < H_HEADS; ++h) {
      sp[h] += __shfl_xor(sp[h], off, 32);
      uu[h] += __shfl_xor(uu[h], off, 32);
    }
  }
  if (lane < H_HEADS) {
    int h = lane;
    scores[((h * N_PTS) + n) * N_PTS + m] =
        (uu[h] + sp[h] * (1.f / (float)KNN)) * INV_SQRT_CH;
  }
}

// softmax over m per (h,n) row; emit f16 for att@V WMMA
__global__ void __launch_bounds__(256) softmax_kernel(const float* __restrict__ scores,
                                                      _Float16* __restrict__ atth)
{
  __shared__ float sd[256];
  int row = blockIdx.x;            // h*256 + n
  int t = threadIdx.x;             // m
  float s = scores[row * N_PTS + t];
  sd[t] = s; __syncthreads();
  for (int o = 128; o > 0; o >>= 1) { if (t < o) sd[t] = fmaxf(sd[t], sd[t + o]); __syncthreads(); }
  float mx = sd[0]; __syncthreads();
  float e = __expf(s - mx);
  sd[t] = e; __syncthreads();
  for (int o = 128; o > 0; o >>= 1) { if (t < o) sd[t] += sd[t + o]; __syncthreads(); }
  atth[row * N_PTS + t] = (_Float16)(e / sd[0]);
}

// vT[h][cc][m] = v[m][h*32+cc]  (f16, K-contiguous for att@V GEMM B)
__global__ void vtrans_kernel(const float* __restrict__ v, _Float16* __restrict__ vT)
{
  int g = blockIdx.x * blockDim.x + threadIdx.x;
  if (g >= H_HEADS * CH * N_PTS) return;
  int m = g & 255;
  int r = g >> 8;
  int cc = r & 31;
  int h  = r >> 5;
  vT[g] = (_Float16)v[m * C_DIM + h * CH + cc];
}

// residual + LayerNorm over C, transpose to (C,N) output
__global__ void __launch_bounds__(128) final_ln_kernel(
    const float* __restrict__ x3, const float* __restrict__ x4,
    const float* __restrict__ g, const float* __restrict__ b,
    float* __restrict__ out)
{
  __shared__ float s1[128], s2[128];
  int n = blockIdx.x, c = threadIdx.x;
  float t = x3[n * C_DIM + c] + x4[n * C_DIM + c];
  s1[c] = t; s2[c] = t * t; __syncthreads();
  for (int o = 64; o > 0; o >>= 1) {
    if (c < o) { s1[c] += s1[c + o]; s2[c] += s2[c + o]; }
    __syncthreads();
  }
  float mu  = s1[0] / (float)C_DIM;
  float var = s2[0] / (float)C_DIM - mu * mu;
  out[c * N_PTS + n] = (t - mu) * rsqrtf(var + EPS_F) * g[c] + b[c];
}

// ---------------------------------------------------------------------------
extern "C" void kernel_launch(void* const* d_in, const int* in_sizes, int n_in,
                              void* d_out, int out_size, void* d_ws, size_t ws_size,
                              hipStream_t stream)
{
  (void)in_sizes; (void)n_in; (void)out_size; (void)ws_size;

  const float* coords   = (const float*)d_in[0];
  const float* features = (const float*)d_in[1];
  const float* conv1_w  = (const float*)d_in[2];
  const float* conv2_w  = (const float*)d_in[3];
  const float* conv3_w  = (const float*)d_in[4];
  const float* proj1_w  = (const float*)d_in[5];
  const float* proj1_b  = (const float*)d_in[6];
  const float* wq = (const float*)d_in[7];  const float* bq = (const float*)d_in[8];
  const float* wk = (const float*)d_in[9];  const float* bk = (const float*)d_in[10];
  const float* wv = (const float*)d_in[11]; const float* bv = (const float*)d_in[12];
  const float* wp = (const float*)d_in[13]; const float* bp = (const float*)d_in[14];
  const float* proj2_w = (const float*)d_in[15]; const float* proj2_b = (const float*)d_in[16];
  const float* ln_g = (const float*)d_in[17]; const float* ln_b = (const float*)d_in[18];
  float* out = (float*)d_out;

  size_t off = 0;
  auto alloc = [&](size_t bytes) -> void* {
    void* p = (char*)d_ws + off;
    off += (bytes + 255) & ~(size_t)255;
    return p;
  };

  int*      idx   = (int*)alloc(P_ROWS * sizeof(int));
  _Float16* Agf   = (_Float16*)alloc((size_t)P_ROWS * 256 * 2);     // A1, reused as A2
  float*    ybuf  = (float*)alloc((size_t)P_ROWS * 256 * 4);        // y1 (..x128) then y2 (x256)
  _Float16* W1h   = (_Float16*)alloc(128 * 256 * 2);
  _Float16* W2h   = (_Float16*)alloc(256 * 256 * 2);
  _Float16* W3h   = (_Float16*)alloc(128 * 512 * 2);
  _Float16* p1h   = (_Float16*)alloc(128 * 128 * 2);
  _Float16* wqT   = (_Float16*)alloc(128 * 128 * 2);
  _Float16* wkT   = (_Float16*)alloc(128 * 128 * 2);
  _Float16* wvT   = (_Float16*)alloc(128 * 128 * 2);
  _Float16* wpT   = (_Float16*)alloc(128 * 128 * 2);
  _Float16* p2T   = (_Float16*)alloc(128 * 128 * 2);
  float*    x1    = (float*)alloc(C_DIM * N_PTS * 4);               // [c][n]
  float*    x2    = (float*)alloc(256 * N_PTS * 4);                 // [c][n]
  _Float16* A3    = (_Float16*)alloc(N_PTS * 512 * 2);
  float*    y3    = (float*)alloc(N_PTS * C_DIM * 4);
  float*    x3    = (float*)alloc(N_PTS * C_DIM * 4);               // [n][c]
  _Float16* x3h   = (_Float16*)alloc(N_PTS * C_DIM * 2);
  float*    qm    = (float*)alloc(N_PTS * C_DIM * 4);
  float*    km    = (float*)alloc(N_PTS * C_DIM * 4);
  float*    vm    = (float*)alloc(N_PTS * C_DIM * 4);
  float*    G     = (float*)alloc(C_DIM * C_DIM * 4);
  float*    gbp   = (float*)alloc(C_DIM * 4);
  float*    rr    = (float*)alloc((size_t)N_PTS * H_HEADS * C_DIM * 4);
  float*    sc    = (float*)alloc((size_t)H_HEADS * N_PTS * N_PTS * 4);
  _Float16* atth  = (_Float16*)alloc((size_t)H_HEADS * N_PTS * N_PTS * 2);
  _Float16* vT    = (_Float16*)alloc((size_t)H_HEADS * CH * N_PTS * 2);
  float*    hid   = (float*)alloc(N_PTS * C_DIM * 4);
  _Float16* hidh  = (_Float16*)alloc(N_PTS * C_DIM * 2);
  float*    x4    = (float*)alloc(N_PTS * C_DIM * 4);

  // simple one-tile-per-wave path (odd shapes)
  auto gemm = [&](const _Float16* A, const _Float16* Bt, const float* bias, float* C,
                  int M, int N, int K, int ldc, int batch, long sA, long sB, long sC) {
    int tiles = (M / 16) * (N / 16);
    dim3 grid((tiles + 7) / 8, batch);
    gemm_f16_wmma<<<grid, 256, 0, stream>>>(A, Bt, bias, C, M, N, K, ldc, sA, sB, sC);
  };
  // blocked path: 128x64 macro-tile, async-staged LDS B panel (K<=256)
  auto gemmB = [&](const _Float16* A, const _Float16* Bt, const float* bias, float* C,
                   int M, int N, int K, int ldc) {
    dim3 grid((M / 128) * (N / 64), 1);
    size_t lds = (size_t)K * 128;   // 64 cols * K halves * 2 bytes
    gemm_f16_wmma_blk<<<grid, 256, lds, stream>>>(A, Bt, bias, C, M, N, K, ldc, 0, 0, 0);
  };
  auto cvt = [&](const float* s, _Float16* d, int n) {
    f2h_kernel<<<(n + 255) / 256, 256, 0, stream>>>(s, d, n);
  };
  auto cvtT = [&](const float* s, _Float16* d, int R, int Cc) {
    f2hT_kernel<<<(R * Cc + 255) / 256, 256, 0, stream>>>(s, d, R, Cc);
  };

  // ---- kNN + weight conversions (conv weights are already [out][in] = Bt) ----
  knn_kernel<<<1, 256, 0, stream>>>(coords, idx);
  cvt(conv1_w, W1h, 128 * 256);
  cvt(conv2_w, W2h, 256 * 256);
  cvt(conv3_w, W3h, 128 * 512);
  cvt(proj1_w, p1h, 128 * 128);
  cvtT(wq, wqT, 128, 128);
  cvtT(wk, wkT, 128, 128);
  cvtT(wv, wvT, 128, 128);
  cvtT(wp, wpT, 128, 128);
  cvtT(proj2_w, p2T, 128, 128);

  // ---- conv1: graph feature -> blocked WMMA GEMM -> inorm/lrelu/max-k ----
  prep_gf_kernel<<<(P_ROWS * 256 + 255) / 256, 256, 0, stream>>>(features, idx, Agf, 128);
  gemmB(Agf, W1h, nullptr, ybuf, P_ROWS, 128, 256, 128);
  inorm_max_kernel<<<128, 256, 0, stream>>>(ybuf, x1, 128);

  // ---- conv2 ----
  prep_gf_kernel<<<(P_ROWS * 256 + 255) / 256, 256, 0, stream>>>(x1, idx, Agf, 128);
  gemmB(Agf, W2h, nullptr, ybuf, P_ROWS, 256, 256, 256);
  inorm_max_kernel<<<256, 256, 0, stream>>>(ybuf, x2, 256);

  // ---- conv3 on [x0,x1,x2] (K=512 -> simple path) ----
  prep_cat_kernel<<<(N_PTS * 512 + 255) / 256, 256, 0, stream>>>(features, x1, x2, A3);
  gemm(A3, W3h, nullptr, y3, N_PTS, 128, 512, 128, 1, 0, 0, 0);
  inorm_nl_kernel<<<128, 256, 0, stream>>>(y3, x3, x3h);

  // ---- q/k/v projections + folded-position GEMM G = proj1_w @ wp ----
  gemmB(x3h, wqT, bq, qm, N_PTS, 128, 128, 128);
  gemmB(x3h, wkT, bk, km, N_PTS, 128, 128, 128);
  gemmB(x3h, wvT, bv, vm, N_PTS, 128, 128, 128);
  gemmB(p1h, wpT, nullptr, G, 128, 128, 128, 128);
  gbp_kernel<<<1, 128, 0, stream>>>(proj1_b, wp, bp, gbp);
  rr_kernel<<<N_PTS, 128, 0, stream>>>(G, qm, rr);

  // ---- fused angular-embedding attention scores + softmax ----
  score_kernel<<<(N_PTS * N_PTS) / 8, 256, 0, stream>>>(coords, idx, qm, km, rr, gbp, sc);
  softmax_kernel<<<H_HEADS * N_PTS, 256, 0, stream>>>(sc, atth);

  // ---- att @ V (batched over heads, N=32 -> simple path), proj2, residual LN ----
  vtrans_kernel<<<(H_HEADS * CH * N_PTS + 255) / 256, 256, 0, stream>>>(vm, vT);
  gemm(atth, vT, nullptr, hid, N_PTS, CH, N_PTS, C_DIM, H_HEADS,
       (long)N_PTS * N_PTS, (long)CH * N_PTS, (long)CH);
  cvt(hid, hidh, N_PTS * C_DIM);
  gemm(hidh, p2T, proj2_b, x4, N_PTS, 128, 128, 128, 1, 0, 0, 0);
  final_ln_kernel<<<N_PTS, 128, 0, stream>>>(x3, x4, ln_g, ln_b, out);
}
